// S5SSM_68109591380649
// MI455X (gfx1250) — matrix operand order repaired and checked by reading, not compiled, not run on hardware
//
#include <hip/hip_runtime.h>
#include <math.h>

// ---------------------------------------------------------------------------
// S5 SSM for MI455X (gfx1250, wave32, WMMA + double-buffered async-to-LDS).
// BSZ=8, H=512, P=128, L=8192.  bf16 WMMA GEMMs + fp32 chunked scan.
// ---------------------------------------------------------------------------

typedef __attribute__((ext_vector_type(16))) __bf16 v16bf;
typedef __attribute__((ext_vector_type(8)))  float  v8f;

#define BSZ    8
#define HDIM   512
#define PDIM   128
#define P2     256         // 2*P (re|im stacked)
#define LSEQ   8192
#define CHUNK  128
#define NCHUNK 64          // LSEQ / CHUNK
#define LTILE  128
#define NLT    64          // LSEQ / LTILE
#define LDAF   132         // f32 LDS row stride, u staging (528 B, 16B aligned)
#define LDAB   40          // bf16 LDS row stride, Xs staging (80 B, 16B aligned)
#define LDBB   40          // bf16 LDS row stride, weight panels (80 B, 16B aligned)
#define LDT    132         // f32 LDS row stride, output transpose

// ---- workspace byte offsets ----
#define WBU_OFF   0UL
#define WC_OFF    (WBU_OFF + (size_t)P2 * HDIM * 2)          // +256 KB
#define LB_OFF    (WC_OFF  + (size_t)HDIM * P2 * 2)          // +256 KB
#define DC_OFF    (LB_OFF  + (size_t)PDIM * 8)
#define BU_OFF    (1UL << 20)
#define XS_OFF    (BU_OFF + (size_t)BSZ * LSEQ * P2 * 4)     // +64 MB
#define CL_OFF    (XS_OFF + (size_t)BSZ * LSEQ * P2 * 2)     // +32 MB
#define CI_OFF    (CL_OFF + (size_t)BSZ * NCHUNK * PDIM * 8) // +512 KB

__device__ __forceinline__ __bf16 f2bf(float x) { return (__bf16)x; }

// 32-bit LDS byte offset of a generic shared-memory pointer.
__device__ __forceinline__ unsigned lds_off(const void* p) {
  return (unsigned)(size_t)p;
}

// Async copy 16B global -> LDS (GLOBAL_LOAD_ASYNC_TO_LDS_B128, ASYNCcnt).
__device__ __forceinline__ void async_copy_b128(unsigned lds_byte_off,
                                                const void* gptr) {
  asm volatile("global_load_async_to_lds_b128 %0, %1, off"
               :
               : "v"(lds_byte_off), "v"(gptr)
               : "memory");
}

__device__ __forceinline__ void wait_async_all() {
  asm volatile("s_wait_asynccnt 0x0" ::: "memory");
}

// ---------------------------------------------------------------------------
// Kernel 1: discretization + weight packing.
//   Wbu[n][h]  n<128: Re(B_bar[p][h]),  n>=128: Im(B_bar[p][h])     (bf16)
//   Wc [h][k]  k<128: 2*Re(C[h][p]),    k>=128: -2*Im(C[h][p])      (bf16)
//   LamBar[p] = exp(Lambda*step), Dc[p] = LamBar^CHUNK              (f32)
// ---------------------------------------------------------------------------
__global__ __launch_bounds__(256) void k_prep(
    const float* __restrict__ Lre, const float* __restrict__ Lim,
    const float* __restrict__ Bw,  const float* __restrict__ Cw,
    const float* __restrict__ logstep, unsigned char* __restrict__ ws) {
  __bf16* Wbu = (__bf16*)(ws + WBU_OFF);
  __bf16* Wc  = (__bf16*)(ws + WC_OFF);
  float2* Lam = (float2*)(ws + LB_OFF);
  float2* Dc  = (float2*)(ws + DC_OFF);
  const int idx = blockIdx.x * 256 + threadIdx.x;   // 0 .. 65535

  { // ---- B_bar weights ----
    const int p = idx >> 9;          // /512
    const int h = idx & (HDIM - 1);
    const float st  = __expf(logstep[p]);
    const float lre = Lre[p], lim = Lim[p];
    const float ea  = __expf(lre * st);
    float sn, cs;
    __sincosf(lim * st, &sn, &cs);
    const float lbr = ea * cs, lbi = ea * sn;          // Lambda_bar
    const float den = lre * lre + lim * lim;
    const float gr  = ((lbr - 1.f) * lre + lbi * lim) / den;  // (Lb-1)/L
    const float gi  = (lbi * lre - (lbr - 1.f) * lim) / den;
    const float br  = Bw[((size_t)p * HDIM + h) * 2 + 0];
    const float bi  = Bw[((size_t)p * HDIM + h) * 2 + 1];
    Wbu[(size_t)p * HDIM + h]          = f2bf(gr * br - gi * bi);
    Wbu[(size_t)(PDIM + p) * HDIM + h] = f2bf(gr * bi + gi * br);
    if (h == 0) {
      Lam[p] = make_float2(lbr, lbi);
      const float eaC = __expf(lre * st * (float)CHUNK);
      float snC, csC;
      __sincosf(lim * st * (float)CHUNK, &snC, &csC);
      Dc[p] = make_float2(eaC * csC, eaC * snC);
    }
  }
  { // ---- C weights (factor 2 and -imag folded in) ----
    const int h = idx >> 7;          // /128
    const int p = idx & (PDIM - 1);
    Wc[(size_t)h * P2 + p]        = f2bf( 2.f * Cw[((size_t)h * PDIM + p) * 2 + 0]);
    Wc[(size_t)h * P2 + PDIM + p] = f2bf(-2.f * Cw[((size_t)h * PDIM + p) * 2 + 1]);
  }
}

// ---------------------------------------------------------------------------
// Kernel 2: Bu[b][l][n] = sum_h u[b][h][l] * Wbu[n][h]   (bf16 WMMA, f32 acc)
// Block: 256 thr (8 waves). Tile M=128 (l) x N=256 (2P), K=512 in 32-chunks.
// Both the u chunk (fp32) and the weight panel (bf16) are async-copied into
// double-buffered LDS; the DMA for chunk kc+1 overlaps the WMMA loop of kc.
// ---------------------------------------------------------------------------
__global__ __launch_bounds__(256) void k_bu(
    const float* __restrict__ u, const __bf16* __restrict__ Wbu,
    float* __restrict__ Bu) {
  __shared__ float  Asf[2][32 * LDAF];     // u panel  [k=h][l]
  __shared__ __bf16 Bsb[2][P2 * LDBB];     // W panel  [n][k]
  const int b     = blockIdx.x / NLT;
  const int lbase = (blockIdx.x % NLT) * LTILE;
  const int tid   = threadIdx.x;
  const int lane  = tid & 31;
  const int w     = tid >> 5;
  const int m     = lane & 15;
  const int hi    = lane >> 4;
  const int row   = w * 16 + m;            // A row (l within tile)
  const int koff  = hi * 16;               // B layout: lanes>=16 hold K=16..31
  const unsigned abase[2] = {lds_off(&Asf[0][0]), lds_off(&Asf[1][0])};
  const unsigned bbase[2] = {lds_off(&Bsb[0][0]), lds_off(&Bsb[1][0])};

  auto stage = [&](int kc, int buf) {
    const int kbase = kc * 32;
    // u chunk [32 h][128 l] fp32 (4x B128 per thread)
    for (int i = tid; i < 32 * 32; i += 256) {
      const int hh  = i >> 5;
      const int seg = i & 31;
      async_copy_b128(abase[buf] + (unsigned)(hh * LDAF + seg * 4) * 4u,
                      u + ((size_t)b * HDIM + kbase + hh) * LSEQ + lbase +
                          seg * 4);
    }
    // weight panel [256 n][32 k] bf16 (4x B128 per thread)
    for (int i = tid; i < P2 * 4; i += 256) {
      const int n   = i >> 2;
      const int seg = i & 3;
      async_copy_b128(bbase[buf] + (unsigned)(n * LDBB + seg * 8) * 2u,
                      Wbu + (size_t)n * HDIM + kbase + seg * 8);
    }
  };

  v8f acc[16];
#pragma unroll
  for (int t = 0; t < 16; ++t)
    acc[t] = (v8f){0.f, 0.f, 0.f, 0.f, 0.f, 0.f, 0.f, 0.f};

  stage(0, 0);
  wait_async_all();
  __syncthreads();

  for (int kc = 0; kc < HDIM / 32; ++kc) {
    const int cur = kc & 1;
    if (kc + 1 < HDIM / 32) stage(kc + 1, cur ^ 1);   // overlap DMA w/ WMMA

    // Build A fragment: pack two fp32 -> bf16 pair per VGPR.
    const float* Ac = &Asf[cur][0];
    union { v16bf v; __bf16 e[16]; } af;
#pragma unroll
    for (int vq = 0; vq < 8; ++vq) {
      const int kv = hi * 16 + 2 * vq;
      af.e[2 * vq]     = f2bf(Ac[kv * LDAF + row]);
      af.e[2 * vq + 1] = f2bf(Ac[(kv + 1) * LDAF + row]);
    }
#pragma unroll
    for (int t = 0; t < 16; ++t) {
      const v16bf bfr =
          *(const v16bf*)&Bsb[cur][(t * 16 + m) * LDBB + koff];
      acc[t] = __builtin_amdgcn_wmma_f32_16x16x32_bf16(
          false, af.v, false, bfr, (short)0, acc[t], false, false);
    }
    wait_async_all();      // next chunk landed
    __syncthreads();       // and everyone is done reading this one
  }

#pragma unroll
  for (int t = 0; t < 16; ++t) {
#pragma unroll
    for (int r = 0; r < 8; ++r) {
      const int l = lbase + w * 16 + hi * 8 + r;   // C/D layout (ISA 7.12.2)
      Bu[((size_t)b * LSEQ + l) * P2 + t * 16 + m] = acc[t][r];
    }
  }
}

// ---------------------------------------------------------------------------
// Kernel 3: local scan inside each 128-step chunk (in place), save chunk tail.
// ---------------------------------------------------------------------------
__global__ __launch_bounds__(128) void k_scan_local(
    float* __restrict__ Bu, float2* __restrict__ clast,
    const float2* __restrict__ Lam) {
  const int b = blockIdx.x / NCHUNK;
  const int c = blockIdx.x % NCHUNK;
  const int p = threadIdx.x;
  const float2 lb = Lam[p];
  float xr = 0.f, xi = 0.f;
  const size_t base = ((size_t)b * LSEQ + (size_t)c * CHUNK) * P2 + p;
  for (int i = 0; i < CHUNK; ++i) {
    const size_t idx = base + (size_t)i * P2;
    const float br = Bu[idx], bi = Bu[idx + PDIM];
    const float nr = lb.x * xr - lb.y * xi + br;
    const float ni = lb.x * xi + lb.y * xr + bi;
    xr = nr; xi = ni;
    Bu[idx] = xr; Bu[idx + PDIM] = xi;
  }
  clast[((size_t)b * NCHUNK + c) * PDIM + p] = make_float2(xr, xi);
}

// ---------------------------------------------------------------------------
// Kernel 4: sequential carry combine across 64 chunks per (b,p).
// ---------------------------------------------------------------------------
__global__ __launch_bounds__(128) void k_scan_carry(
    const float2* __restrict__ clast, float2* __restrict__ cin,
    const float2* __restrict__ Dc) {
  const int gid = blockIdx.x * 128 + threadIdx.x;   // 0..1023
  const int b = gid >> 7;
  const int p = gid & (PDIM - 1);
  const float2 d = Dc[p];
  float cr = 0.f, ci = 0.f;
  for (int c = 0; c < NCHUNK; ++c) {
    const size_t idx = ((size_t)b * NCHUNK + c) * PDIM + p;
    cin[idx] = make_float2(cr, ci);                 // incoming state of chunk
    const float2 lst = clast[idx];
    const float nr = d.x * cr - d.y * ci + lst.x;
    const float ni = d.x * ci + d.y * cr + lst.y;
    cr = nr; ci = ni;
  }
}

// ---------------------------------------------------------------------------
// Kernel 5: fixup x_t = local_t + Lb^(i+1)*carry, emit bf16 [xr|xi] for GEMM2.
// ---------------------------------------------------------------------------
__global__ __launch_bounds__(128) void k_scan_fix(
    const float* __restrict__ Bu, const float2* __restrict__ cin,
    const float2* __restrict__ Lam, __bf16* __restrict__ Xs) {
  const int b = blockIdx.x / NCHUNK;
  const int c = blockIdx.x % NCHUNK;
  const int p = threadIdx.x;
  const float2 lb = Lam[p];
  const float2 c0 = cin[((size_t)b * NCHUNK + c) * PDIM + p];
  float cr = c0.x, ci = c0.y;
  const size_t base = ((size_t)b * LSEQ + (size_t)c * CHUNK) * P2 + p;
  for (int i = 0; i < CHUNK; ++i) {
    const float nr = lb.x * cr - lb.y * ci;
    const float ni = lb.x * ci + lb.y * cr;
    cr = nr; ci = ni;
    const size_t idx = base + (size_t)i * P2;
    Xs[idx]        = f2bf(Bu[idx] + cr);
    Xs[idx + PDIM] = f2bf(Bu[idx + PDIM] + ci);
  }
}

// ---------------------------------------------------------------------------
// Kernel 6: ys = xs2 * Wc + D*u, exact GELU, output in [b][h][l] layout.
// A (bf16 Xs) and weight panel async-copied into double-buffered LDS.
// Epilogue transposes tiles through LDS for coalesced D*u reads and stores.
// ---------------------------------------------------------------------------
__global__ __launch_bounds__(256) void k_out(
    const __bf16* __restrict__ Xs, const __bf16* __restrict__ Wc,
    const float* __restrict__ u, const float* __restrict__ Dvec,
    float* __restrict__ out) {
  __shared__ __bf16 Asb[2][LTILE * LDAB];   // Xs panel [l][k]
  __shared__ __bf16 Bsb[2][P2 * LDBB];      // Wc panel [h][k]
  __shared__ float  T[32 * LDT];
  const int b     = blockIdx.x >> 7;        // 128 blocks per batch
  const int rem   = blockIdx.x & 127;
  const int lbase = (rem >> 1) * LTILE;
  const int hbase = (rem & 1) * 256;
  const int tid   = threadIdx.x;
  const int lane  = tid & 31;
  const int w     = tid >> 5;
  const int m     = lane & 15;
  const int hi    = lane >> 4;
  const int koff  = hi * 16;
  const unsigned abase[2] = {lds_off(&Asb[0][0]), lds_off(&Asb[1][0])};
  const unsigned bbase[2] = {lds_off(&Bsb[0][0]), lds_off(&Bsb[1][0])};

  auto stage = [&](int kc, int buf) {
    const int kbase = kc * 32;
    // Xs chunk [128 l][32 k] bf16 (2x B128 per thread)
    for (int i = tid; i < LTILE * 4; i += 256) {
      const int ll  = i >> 2;
      const int seg = i & 3;
      async_copy_b128(abase[buf] + (unsigned)(ll * LDAB + seg * 8) * 2u,
                      Xs + ((size_t)b * LSEQ + lbase + ll) * P2 + kbase +
                          seg * 8);
    }
    // Wc panel [256 h][32 k] bf16 (4x B128 per thread)
    for (int i = tid; i < P2 * 4; i += 256) {
      const int n   = i >> 2;
      const int seg = i & 3;
      async_copy_b128(bbase[buf] + (unsigned)(n * LDBB + seg * 8) * 2u,
                      Wc + (size_t)(hbase + n) * P2 + kbase + seg * 8);
    }
  };

  v8f acc[16];
#pragma unroll
  for (int t = 0; t < 16; ++t)
    acc[t] = (v8f){0.f, 0.f, 0.f, 0.f, 0.f, 0.f, 0.f, 0.f};

  stage(0, 0);
  wait_async_all();
  __syncthreads();

  for (int kc = 0; kc < P2 / 32; ++kc) {
    const int cur = kc & 1;
    if (kc + 1 < P2 / 32) stage(kc + 1, cur ^ 1);

    const unsigned int* r32 =
        (const unsigned int*)&Asb[cur][(w * 16 + m) * LDAB];
    union { v16bf v; unsigned int q[8]; } af;
#pragma unroll
    for (int vq = 0; vq < 8; ++vq) {
      const int kv = hi * 16 + 2 * vq;
      af.q[vq] = r32[kv >> 1];
    }
#pragma unroll
    for (int t = 0; t < 16; ++t) {
      const v16bf bfr =
          *(const v16bf*)&Bsb[cur][(t * 16 + m) * LDBB + koff];
      acc[t] = __builtin_amdgcn_wmma_f32_16x16x32_bf16(
          false, af.v, false, bfr, (short)0, acc[t], false, false);
    }
    wait_async_all();
    __syncthreads();
  }

  // Epilogue: per 32-wide h chunk, transpose (l,h)->(h,l) via LDS.
  for (int hc = 0; hc < 8; ++hc) {
    __syncthreads();
#pragma unroll
    for (int t2 = 0; t2 < 2; ++t2) {
      const int t = hc * 2 + t2;
#pragma unroll
      for (int r = 0; r < 8; ++r)
        T[(t2 * 16 + m) * LDT + w * 16 + hi * 8 + r] = acc[t][r];
    }
    __syncthreads();
    const int hl   = tid >> 3;
    const int lsub = tid & 7;
    const int h    = hbase + hc * 32 + hl;
    const float dh = Dvec[h];
#pragma unroll
    for (int j = 0; j < 16; ++j) {
      const int l  = lsub + j * 8;
      const int gl = lbase + l;
      const float uu = u[((size_t)b * HDIM + h) * LSEQ + gl];
      const float y  = T[hl * LDT + l] + dh * uu;
      const float g  = 0.5f * y * (1.f + erff(y * 0.70710678118654752f));
      out[((size_t)b * HDIM + h) * LSEQ + gl] = g;
    }
  }
}

// ---------------------------------------------------------------------------
extern "C" void kernel_launch(void* const* d_in, const int* in_sizes, int n_in,
                              void* d_out, int out_size, void* d_ws,
                              size_t ws_size, hipStream_t stream) {
  (void)in_sizes; (void)n_in; (void)out_size; (void)ws_size;
  const float* u    = (const float*)d_in[0];  // (8,1,512,1,8192) == [b][h][l]
  const float* Lre  = (const float*)d_in[1];
  const float* Lim  = (const float*)d_in[2];
  const float* Bw   = (const float*)d_in[3];  // (P,H,2)
  const float* Cw   = (const float*)d_in[4];  // (H,P,2)
  const float* Dv   = (const float*)d_in[5];  // (H)
  const float* lstp = (const float*)d_in[6];  // (P,1)
  float* out        = (float*)d_out;

  unsigned char* ws = (unsigned char*)d_ws;
  __bf16* Wbu  = (__bf16*)(ws + WBU_OFF);
  __bf16* Wc   = (__bf16*)(ws + WC_OFF);
  float2* Lam  = (float2*)(ws + LB_OFF);
  float2* Dc   = (float2*)(ws + DC_OFF);
  float*  Bu   = (float*)(ws + BU_OFF);
  __bf16* Xs   = (__bf16*)(ws + XS_OFF);
  float2* Clst = (float2*)(ws + CL_OFF);
  float2* Cin  = (float2*)(ws + CI_OFF);

  k_prep<<<256, 256, 0, stream>>>(Lre, Lim, Bw, Cw, lstp, ws);
  k_bu<<<BSZ * NLT, 256, 0, stream>>>(u, Wbu, Bu);
  k_scan_local<<<BSZ * NCHUNK, 128, 0, stream>>>(Bu, Clst, Lam);
  k_scan_carry<<<BSZ, 128, 0, stream>>>(Clst, Cin, Dc);
  k_scan_fix<<<BSZ * NCHUNK, 128, 0, stream>>>(Bu, Cin, Lam, Xs);
  k_out<<<BSZ * 128, 256, 0, stream>>>(Xs, Wc, u, Dv, out);
}
